// Kenn_30674656428437
// MI455X (gfx1250) — matrix operand order repaired and checked by reading, not compiled, not run on hardware
//
#include <hip/hip_runtime.h>
#include <math.h>

#define D_IN   2048
#define H_DIM  50
#define HP     64      // hidden padded to 64 cols
#define C_CLS  6
#define ZSTR   8       // z row stride (floats), padded for float4 gathers
#define TROWS  128     // rows per block in MLP kernel
#define KSTEPS (D_IN / 32)

typedef __attribute__((ext_vector_type(16))) __bf16 v16bf;
typedef __attribute__((ext_vector_type(8)))  float  v8f;
typedef unsigned int u32x4 __attribute__((ext_vector_type(4)));

union Frag16 {
    v16bf v;
    u32x4 q[2];
    unsigned int u[8];
};

// ---- f32 -> bf16 helpers ----
__device__ __forceinline__ unsigned short bf16_1(float a) {
    unsigned int u = __float_as_uint(a);
    u += 0x7FFFu + ((u >> 16) & 1u);
    return (unsigned short)(u >> 16);
}

#if __has_builtin(__builtin_amdgcn_cvt_pk_bf16_f32)
typedef __attribute__((ext_vector_type(2))) __bf16 v2bf;
__device__ __forceinline__ unsigned int bf16_2(float a, float b) {
    v2bf r = __builtin_amdgcn_cvt_pk_bf16_f32(a, b);
    return __builtin_bit_cast(unsigned int, r);
}
#elif __has_builtin(__builtin_amdgcn_perm)
// round-half-up bias then one v_perm_b32 to gather the two high halves:
// dst = { (b+0x8000)[31:16], (a+0x8000)[31:16] }  (3 VALU ops per pair)
__device__ __forceinline__ unsigned int bf16_2(float a, float b) {
    unsigned int ra = __float_as_uint(a) + 0x8000u;
    unsigned int rb = __float_as_uint(b) + 0x8000u;
    return __builtin_amdgcn_perm(rb, ra, 0x07060302u);
}
#else
__device__ __forceinline__ unsigned int bf16_2(float a, float b) {
    unsigned int ua = __float_as_uint(a);
    ua += 0x7FFFu + ((ua >> 16) & 1u);
    unsigned int ub = __float_as_uint(b);
    ub += 0x7FFFu + ((ub >> 16) & 1u);
    return (ua >> 16) | (ub & 0xFFFF0000u);
}
#endif

// ---- async global->LDS staging (CDNA5) with plain-load fallback ----
#if __has_builtin(__builtin_amdgcn_global_load_async_to_lds_b128)
#define ASYNC_LDS 1
typedef int v4i __attribute__((vector_size(4 * sizeof(int))));
typedef __attribute__((address_space(1))) v4i* gp_v4i;   // global b128 ptr
typedef __attribute__((address_space(3))) v4i* lp_v4i;   // LDS b128 ptr (32-bit)

__device__ __forceinline__ void async_b128(const void* g, void* l) {
    // integer round-trips avoid disallowed generic->AS pointer casts;
    // generic LDS addresses carry the LDS byte offset in the low 32 bits.
    gp_v4i gp = (gp_v4i)(unsigned long long)g;
    lp_v4i lp = (lp_v4i)(unsigned int)(unsigned long long)l;
    __builtin_amdgcn_global_load_async_to_lds_b128(gp, lp, 0, 0);
}
__device__ __forceinline__ void wait_async0() {
#if __has_builtin(__builtin_amdgcn_s_wait_asynccnt)
    __builtin_amdgcn_s_wait_asynccnt(0);
#else
    asm volatile("s_wait_asynccnt 0x0" ::: "memory");
#endif
}
#else
#define ASYNC_LDS 0
#endif

// Pack 4x float4 (16 f32, 16 f32 in consecutive K order, same order as loadB) into a v16bf frag
__device__ __forceinline__ v16bf packA(float4 f0, float4 f1, float4 f2, float4 f3) {
    Frag16 f;
    f.u[0] = bf16_2(f0.x, f0.y);
    f.u[1] = bf16_2(f0.z, f0.w);
    f.u[2] = bf16_2(f1.x, f1.y);
    f.u[3] = bf16_2(f1.z, f1.w);
    f.u[4] = bf16_2(f2.x, f2.y);
    f.u[5] = bf16_2(f2.z, f2.w);
    f.u[6] = bf16_2(f3.x, f3.y);
    f.u[7] = bf16_2(f3.z, f3.w);
    return f.v;
}

// A fragment (16x32, M=lane&15): half=0 -> K=kk+0..15; half=1 -> K=kk+16..31
__device__ __forceinline__ v16bf loadA_f32row(const float* row, int half, int kk) {
    const float4* p = (const float4*)(row + kk);
    return packA(p[half * 4], p[half * 4 + 1], p[half * 4 + 2], p[half * 4 + 3]);
}

// B fragment (32x16): lane holds column n; lanes 0-15 K=kk+0..15, lanes 16-31 K=kk+16..31.
// WT is bf16, n-major with row stride `stride` elements -> two contiguous 16B loads.
__device__ __forceinline__ v16bf loadB(const unsigned short* WT, int stride, int n, int half, int kk) {
    const u32x4* p = (const u32x4*)(WT + (size_t)n * stride + kk + half * 16);
    Frag16 f;
    f.q[0] = p[0];
    f.q[1] = p[1];
    return f.v;
}

// ------------------------------------------------------------------
// Prep: transpose+pad weights to bf16 (n-major): W1T[HP][D_IN], W2T/W3T[HP][HP], W4T[16][HP]
// ------------------------------------------------------------------
__global__ void kenn_prep_weights(const float* __restrict__ W1, const float* __restrict__ W2,
                                  const float* __restrict__ W3, const float* __restrict__ W4,
                                  unsigned short* __restrict__ W1T, unsigned short* __restrict__ W2T,
                                  unsigned short* __restrict__ W3T, unsigned short* __restrict__ W4T) {
    const int t1 = HP * D_IN;
    const int t2 = HP * HP;
    const int t4 = 16 * HP;
    int idx = blockIdx.x * blockDim.x + threadIdx.x;
    if (idx < t1) {
        int n = idx / D_IN, k = idx % D_IN;
        float v = (n < H_DIM) ? W1[(size_t)k * H_DIM + n] : 0.f;
        W1T[idx] = bf16_1(v);
    } else if (idx < t1 + t2) {
        int j = idx - t1;
        int n = j / HP, k = j % HP;
        float v = (n < H_DIM && k < H_DIM) ? W2[k * H_DIM + n] : 0.f;
        W2T[j] = bf16_1(v);
    } else if (idx < t1 + 2 * t2) {
        int j = idx - t1 - t2;
        int n = j / HP, k = j % HP;
        float v = (n < H_DIM && k < H_DIM) ? W3[k * H_DIM + n] : 0.f;
        W3T[j] = bf16_1(v);
    } else if (idx < t1 + 2 * t2 + t4) {
        int j = idx - t1 - 2 * t2;
        int n = j / HP, k = j % HP;
        float v = (n < C_CLS && k < H_DIM) ? W4[k * C_CLS + n] : 0.f;
        W4T[j] = bf16_1(v);
    }
}

// ------------------------------------------------------------------
// Fused MLP: z = (relu(relu(relu(X@W1+b1)@W2+b2)@W3+b3))@W4 + b4
// 256 threads = 8 waves; each wave owns 16 rows; WMMA bf16 everywhere.
// Layer-1 weights double-buffered in LDS via async copies (CDNA5 path).
// ------------------------------------------------------------------
__global__ __launch_bounds__(256)
void kenn_mlp(const float* __restrict__ features,
              const unsigned short* __restrict__ W1T, const unsigned short* __restrict__ W2T,
              const unsigned short* __restrict__ W3T, const unsigned short* __restrict__ W4T,
              const float* __restrict__ b1, const float* __restrict__ b2,
              const float* __restrict__ b3, const float* __restrict__ b4,
              float* __restrict__ z, int N) {
    __shared__ __align__(16) float xA[TROWS][HP];
    __shared__ __align__(16) float xB[TROWS][HP];
    __shared__ __align__(16) unsigned short w2lds[HP * HP];
    __shared__ __align__(16) unsigned short w3lds[HP * HP];
    __shared__ __align__(16) unsigned short w4lds[16 * HP];
#if ASYNC_LDS
    __shared__ __align__(16) unsigned short w1buf[2][32 * HP];   // 2 x 4KB K-chunks
#endif

    const int tid  = threadIdx.x;
    const int wave = tid >> 5;
    const int lane = tid & 31;
    const int half = lane >> 4;
    const int m    = lane & 15;
    const int rowBase = blockIdx.x * TROWS + wave * 16;
    const int rowG    = rowBase + m;
    const int rowClamped = rowG < N ? rowG : (N - 1);

    const v8f vzero = {};
    v8f acc[4];
#pragma unroll
    for (int t = 0; t < 4; ++t) acc[t] = vzero;

#if ASYNC_LDS
    // stage small-layer weights (waited on together with W1 chunk 0 at layer-1 step 0)
    for (int o = tid * 8; o < HP * HP; o += 256 * 8) {
        async_b128(W2T + o, w2lds + o);
        async_b128(W3T + o, w3lds + o);
    }
    for (int o = tid * 8; o < 16 * HP; o += 256 * 8) async_b128(W4T + o, w4lds + o);
    // stage W1 chunk 0: thread copies 16B; chunk layout [n][32] bf16
    {
        int n = tid >> 2, j = (tid & 3) * 8;
        async_b128(W1T + (size_t)n * D_IN + 0 + j, &w1buf[0][n * 32 + j]);
    }
#endif

    // ---- Layer 1: K = 2048, memory-bound stream of features ----
    const float* arow = features + (size_t)rowClamped * D_IN;
    for (int i = 0; i < KSTEPS; ++i) {
        const int kk = i * 32;
#if ASYNC_LDS
        wait_async0();           // my copy of chunk i (and initial stages) done
        __syncthreads();         // everyone's copies done; prev compute done before overwrite
        if (i + 1 < KSTEPS) {    // prefetch chunk i+1 into the other buffer
            int n = tid >> 2, j = (tid & 3) * 8;
            async_b128(W1T + (size_t)n * D_IN + (kk + 32) + j, &w1buf[(i + 1) & 1][n * 32 + j]);
        }
        const unsigned short* bsrc = &w1buf[i & 1][0];
        const int bstride = 32, bkk = 0;
#else
        const unsigned short* bsrc = W1T;
        const int bstride = D_IN, bkk = kk;
#endif
        __builtin_prefetch(arow + kk + 256, 0, 3);
        v16bf a = loadA_f32row(arow, half, kk);
#pragma unroll
        for (int t = 0; t < 4; ++t) {
            v16bf b = loadB(bsrc, bstride, t * 16 + m, half, bkk);
            acc[t] = __builtin_amdgcn_wmma_f32_16x16x32_bf16(false, a, false, b,
                                                             (short)0, acc[t], false, false);
        }
    }
#if !ASYNC_LDS
    // fallback: copy small-layer weights to LDS with plain loads
    for (int o = tid; o < HP * HP; o += 256) { w2lds[o] = W2T[o]; w3lds[o] = W3T[o]; }
    for (int o = tid; o < 16 * HP; o += 256) w4lds[o] = W4T[o];
#endif
#pragma unroll
    for (int t = 0; t < 4; ++t) {
        int col = t * 16 + m;
        float bias = (col < H_DIM) ? b1[col] : 0.f;
#pragma unroll
        for (int i = 0; i < 8; ++i) {
            int r = wave * 16 + half * 8 + i;
            float v = acc[t][i] + bias;
            xA[r][col] = v > 0.f ? v : 0.f;
        }
    }
    __syncthreads();

    // ---- Layer 2: xA @ W2 -> xB ----
#pragma unroll
    for (int t = 0; t < 4; ++t) acc[t] = vzero;
#pragma unroll
    for (int kk = 0; kk < HP; kk += 32) {
        v16bf a = loadA_f32row(&xA[wave * 16 + m][0], half, kk);
#pragma unroll
        for (int t = 0; t < 4; ++t) {
            v16bf b = loadB(w2lds, HP, t * 16 + m, half, kk);
            acc[t] = __builtin_amdgcn_wmma_f32_16x16x32_bf16(false, a, false, b,
                                                             (short)0, acc[t], false, false);
        }
    }
#pragma unroll
    for (int t = 0; t < 4; ++t) {
        int col = t * 16 + m;
        float bias = (col < H_DIM) ? b2[col] : 0.f;
#pragma unroll
        for (int i = 0; i < 8; ++i) {
            int r = wave * 16 + half * 8 + i;
            float v = acc[t][i] + bias;
            xB[r][col] = v > 0.f ? v : 0.f;
        }
    }
    __syncthreads();

    // ---- Layer 3: xB @ W3 -> xA ----
#pragma unroll
    for (int t = 0; t < 4; ++t) acc[t] = vzero;
#pragma unroll
    for (int kk = 0; kk < HP; kk += 32) {
        v16bf a = loadA_f32row(&xB[wave * 16 + m][0], half, kk);
#pragma unroll
        for (int t = 0; t < 4; ++t) {
            v16bf b = loadB(w3lds, HP, t * 16 + m, half, kk);
            acc[t] = __builtin_amdgcn_wmma_f32_16x16x32_bf16(false, a, false, b,
                                                             (short)0, acc[t], false, false);
        }
    }
    __syncthreads();
#pragma unroll
    for (int t = 0; t < 4; ++t) {
        int col = t * 16 + m;
        float bias = (col < H_DIM) ? b3[col] : 0.f;
#pragma unroll
        for (int i = 0; i < 8; ++i) {
            int r = wave * 16 + half * 8 + i;
            float v = acc[t][i] + bias;
            xA[r][col] = v > 0.f ? v : 0.f;
        }
    }
    __syncthreads();

    // ---- Layer 4: xA @ W4 -> z (6 valid cols; one 16-col N tile) ----
    v8f zc = vzero;
#pragma unroll
    for (int kk = 0; kk < HP; kk += 32) {
        v16bf a = loadA_f32row(&xA[wave * 16 + m][0], half, kk);
        v16bf b = loadB(w4lds, HP, m, half, kk);
        zc = __builtin_amdgcn_wmma_f32_16x16x32_bf16(false, a, false, b,
                                                     (short)0, zc, false, false);
    }
    {
        int col = m;
#pragma unroll
        for (int i = 0; i < 8; ++i) {
            int rg = blockIdx.x * TROWS + wave * 16 + half * 8 + i;
            if (rg < N) {
                if (col < C_CLS)      z[(size_t)rg * ZSTR + col] = zc[i] + b4[col];
                else if (col < ZSTR)  z[(size_t)rg * ZSTR + col] = 0.f;
            }
        }
    }
}

// ------------------------------------------------------------------
// KENN relational layer kernels
// ------------------------------------------------------------------
__global__ void kenn_zero(float* __restrict__ p, int n) {
    int i = blockIdx.x * blockDim.x + threadIdx.x;
    if (i < n) p[i] = 0.f;
}

__global__ __launch_bounds__(256)
void kenn_edge(const float* __restrict__ z, const float* __restrict__ rel,
               const int* __restrict__ sx, const int* __restrict__ sy,
               const float* __restrict__ cw, float* __restrict__ dz, int M) {
    int e = blockIdx.x * blockDim.x + threadIdx.x;
    if (e >= M) return;
    const int ix = sx[e];
    const int iy = sy[e];
    const float r = rel[e];
    const float4* zx4 = (const float4*)(z + (size_t)ix * ZSTR);
    const float4* zy4 = (const float4*)(z + (size_t)iy * ZSTR);
    float4 zx0 = zx4[0], zx1 = zx4[1];
    float4 zy0 = zy4[0], zy1 = zy4[1];
    float zxv[C_CLS] = {zx0.x, zx0.y, zx0.z, zx0.w, zx1.x, zx1.y};
    float zyv[C_CLS] = {zy0.x, zy0.y, zy0.z, zy0.w, zy1.x, zy1.y};
#pragma unroll
    for (int c = 0; c < C_CLS; ++c) {
        float a0 = -zxv[c];
        float a1 = -r;
        float a2 = zyv[c];
        float mx = fmaxf(a0, fmaxf(a1, a2));
        float e0 = __expf(a0 - mx);
        float e1 = __expf(a1 - mx);
        float e2 = __expf(a2 - mx);
        float inv = 1.0f / (e0 + e1 + e2);
        float w = cw[c];
        atomicAdd(&dz[(size_t)ix * ZSTR + c], -w * e0 * inv); // sign -1 at x
        atomicAdd(&dz[(size_t)iy * ZSTR + c],  w * e2 * inv); // sign +1 at y
    }
}

__global__ void kenn_update(float* __restrict__ z, float* __restrict__ dz, int n) {
    int i = blockIdx.x * blockDim.x + threadIdx.x;
    if (i < n) {
        z[i] += dz[i];
        dz[i] = 0.f; // reset for next layer
    }
}

__global__ void kenn_softmax_out(const float* __restrict__ z, float* __restrict__ out, int N) {
    int n = blockIdx.x * blockDim.x + threadIdx.x;
    if (n >= N) return;
    const float4* z4 = (const float4*)(z + (size_t)n * ZSTR);
    float4 a = z4[0], b = z4[1];
    float v[C_CLS] = {a.x, a.y, a.z, a.w, b.x, b.y};
    float mx = v[0];
#pragma unroll
    for (int c = 1; c < C_CLS; ++c) mx = fmaxf(mx, v[c]);
    float s = 0.f;
    float e[C_CLS];
#pragma unroll
    for (int c = 0; c < C_CLS; ++c) { e[c] = __expf(v[c] - mx); s += e[c]; }
    float inv = 1.0f / s;
#pragma unroll
    for (int c = 0; c < C_CLS; ++c) out[(size_t)n * C_CLS + c] = e[c] * inv;
}

// ------------------------------------------------------------------
extern "C" void kernel_launch(void* const* d_in, const int* in_sizes, int n_in,
                              void* d_out, int out_size, void* d_ws, size_t ws_size,
                              hipStream_t stream) {
    const float* features  = (const float*)d_in[0];
    const float* relations = (const float*)d_in[1];
    const int*   sx        = (const int*)d_in[2];
    const int*   sy        = (const int*)d_in[3];
    const float* W1 = (const float*)d_in[4];
    const float* b1 = (const float*)d_in[5];
    const float* W2 = (const float*)d_in[6];
    const float* b2 = (const float*)d_in[7];
    const float* W3 = (const float*)d_in[8];
    const float* b3 = (const float*)d_in[9];
    const float* W4 = (const float*)d_in[10];
    const float* b4 = (const float*)d_in[11];
    const float* cw = (const float*)d_in[12];

    const int N = in_sizes[0] / D_IN;
    const int M = in_sizes[2];
    const int L = in_sizes[12] / C_CLS;

    char* ws = (char*)d_ws;
    size_t off = 0;
    auto carve = [&](size_t bytes) -> char* {
        char* p = ws + off;
        off += (bytes + 511) & ~(size_t)511;
        return p;
    };
    unsigned short* W1T = (unsigned short*)carve((size_t)HP * D_IN * 2);
    unsigned short* W2T = (unsigned short*)carve((size_t)HP * HP * 2);
    unsigned short* W3T = (unsigned short*)carve((size_t)HP * HP * 2);
    unsigned short* W4T = (unsigned short*)carve((size_t)16 * HP * 2);
    float* z  = (float*)carve((size_t)N * ZSTR * 4);
    float* dz = (float*)carve((size_t)N * ZSTR * 4);

    {
        int total = HP * D_IN + 2 * HP * HP + 16 * HP;
        kenn_prep_weights<<<(total + 255) / 256, 256, 0, stream>>>(W1, W2, W3, W4,
                                                                   W1T, W2T, W3T, W4T);
    }
    kenn_mlp<<<(N + TROWS - 1) / TROWS, 256, 0, stream>>>(features, W1T, W2T, W3T, W4T,
                                                          b1, b2, b3, b4, z, N);
    const int NZ = N * ZSTR;
    kenn_zero<<<(NZ + 255) / 256, 256, 0, stream>>>(dz, NZ);
    for (int l = 0; l < L; ++l) {
        kenn_edge<<<(M + 255) / 256, 256, 0, stream>>>(z, relations, sx, sy,
                                                       cw + l * C_CLS, dz, M);
        kenn_update<<<(NZ + 255) / 256, 256, 0, stream>>>(z, dz, NZ);
    }
    kenn_softmax_out<<<(N + 255) / 256, 256, 0, stream>>>(z, (float*)d_out, N);
}
